// GTCN_V1_39367670235765
// MI455X (gfx1250) — matrix operand order repaired and checked by reading, not compiled, not run on hardware
//
#include <hip/hip_runtime.h>

typedef __bf16 bhalf;
typedef __attribute__((ext_vector_type(16))) __bf16 v16bf;
typedef __attribute__((ext_vector_type(4)))  __bf16 v4bf;
typedef __attribute__((ext_vector_type(8)))  float  f32x8;

namespace {
constexpr int Bc = 64, Cc = 3, Tc = 512, Vc = 25, HD = 64, Ec = 48, NCc = 60;
constexpr int MROWS = Bc * Tc * Vc;       // 819200 rows of the GCN GEMMs
constexpr float EPS = 1e-5f;
}

// -------- fragment-order permutation helpers (wave32 layouts, CDNA5 ISA) -----
// A tile element (row r in 128-row tile, k col c) -> flat index in permuted LDS.
// Region = (rowTile * KC + kChunk) of 512 bhalf (32 lanes x 16 elems).
// For c aligned to 4, the 4 consecutive k's land in 4 contiguous bhalf slots.
__device__ __forceinline__ int a_perm_idx(int r, int c, int KC) {
  int rowTile = r >> 4, m = r & 15;
  int kc = c >> 5, kp = c & 31;
  int half = (kp >> 3) & 1;
  int e = (kp & 7) | ((kp >> 4) << 3);
  return ((rowTile * KC + kc) * 32 + (m + 16 * half)) * 16 + e;
}
// B tile element (k row K, col n); regions laid out (kChunk * NT + nTile).
__device__ __forceinline__ int b_perm_idx(int K, int n, int NT) {
  int kc = K >> 5, kp = K & 31;
  int half = (kp >> 3) & 1, e = (kp & 7) | ((kp >> 4) << 3);
  int nt = n >> 4;
  return ((kc * NT + nt) * 32 + ((n & 15) + 16 * half)) * 16 + e;
}

__device__ __forceinline__ void store_pk4(bhalf* dst, float4 x) {
  v4bf p;
  p[0] = (bhalf)x.x; p[1] = (bhalf)x.y; p[2] = (bhalf)x.z; p[3] = (bhalf)x.w;
  *(v4bf*)dst = p;                       // 8-byte aligned packed store
}

// ---------------- small utility kernels ---------------------------------------

__global__ void k_zero(float* p, int n) {
  int i = blockIdx.x * blockDim.x + threadIdx.x;
  if (i < n) p[i] = 0.f;
}

// Adjacency count matrix: Adj[dst][src] = #edges (scatter-add == Adj @ X)
__global__ void k_adj(const int* __restrict__ edge, float* __restrict__ Adj) {
  int tid = threadIdx.x;
  for (int i = tid; i < Vc * Vc; i += 64) Adj[i] = 0.f;
  __syncthreads();
  if (tid < Ec) {
    int src = edge[tid * 2 + 0];
    int dst = edge[tid * 2 + 1];
    atomicAdd(&Adj[dst * Vc + src], 1.0f);
  }
}

// ---------------- GCN dual GEMM: G = X@sW + sb, Mg = X@W ----------------------
// 256 threads = 8 waves; block covers 128 rows x 64 cols of both outputs.
// mode 0: rows gathered from X (B,C,T,V) (layer 1, K=3); mode 1: rows contig (K=64).
__global__ void __launch_bounds__(256)
k_gcn_gemm(const float* Xin, int mode, int K, int Kpad,
           const float* __restrict__ sW, const float* __restrict__ Wm,
           const float* __restrict__ sb, float* G, float* __restrict__ Mg) {
  __shared__ v16bf As4[8 * 2 * 32];    // 128 rows x 64 k, fragment order (16 KB)
  __shared__ v16bf W14[2 * 4 * 32];    // 64 x 64, fragment order (8 KB)
  __shared__ v16bf W24[2 * 4 * 32];
  bhalf* As = (bhalf*)As4;
  bhalf* Ws1 = (bhalf*)W14;
  bhalf* Ws2 = (bhalf*)W24;
  const int tid = threadIdx.x;
  const int row0 = blockIdx.x * 128;

  if (mode == 0) {                      // zero only needed when K < Kpad
    for (int i = tid; i < 4096; i += 256) ((float*)As)[i] = 0.f;
    for (int i = tid; i < 2048; i += 256) { ((float*)Ws1)[i] = 0.f; ((float*)Ws2)[i] = 0.f; }
    __syncthreads();
  }
  for (int i = tid; i < K * 64; i += 256) {
    int r = i >> 6, n = i & 63;                 // weights are (K,64) row-major
    int d = b_perm_idx(r, n, 4);
    Ws1[d] = (bhalf)sW[i];
    Ws2[d] = (bhalf)Wm[i];
  }
  if (mode == 0) {
    for (int i = tid; i < 128 * K; i += 256) {
      int r = i / K, c = i % K;
      int m = row0 + r;
      int b = m / (Tc * Vc); int rem = m % (Tc * Vc);
      int t = rem / Vc; int vv = rem % Vc;
      float v = Xin[(((size_t)b * Cc + c) * Tc + t) * Vc + vv];  // X (B,C,T,V)
      As[a_perm_idx(r, c, 2)] = (bhalf)v;
    }
  } else {                              // K == 64, contiguous rows: float4 loads
    for (int i = tid; i < 128 * 16; i += 256) {
      int r = i >> 4, c4 = (i & 15) << 2;
      float4 x = ((const float4*)(Xin + (size_t)(row0 + r) * HD))[c4 >> 2];
      store_pk4(As + a_perm_idx(r, c4, 2), x);
    }
  }
  __syncthreads();

  const int wave = tid >> 5, lane = tid & 31;
  f32x8 accG[4] = {}, accM[4] = {};

  for (int kk = 0; kk < Kpad; kk += 32) {
    const int kc = kk >> 5;
    v16bf a = As4[(wave * 2 + kc) * 32 + lane];
#pragma unroll
    for (int nt = 0; nt < 4; ++nt) {
      v16bf b1 = W14[(kc * 4 + nt) * 32 + lane];
      accG[nt] = __builtin_amdgcn_wmma_f32_16x16x32_bf16(
          false, a, false, b1, (short)0, accG[nt], false, false);
      v16bf b2 = W24[(kc * 4 + nt) * 32 + lane];
      accM[nt] = __builtin_amdgcn_wmma_f32_16x16x32_bf16(
          false, a, false, b2, (short)0, accM[nt], false, false);
    }
  }

  const int nloc = lane & 15, half = lane >> 4;
  const int gr0 = row0 + wave * 16 + half * 8;
#pragma unroll
  for (int nt = 0; nt < 4; ++nt) {
    int n = nt * 16 + nloc;
    float bias = sb[n];
#pragma unroll
    for (int r = 0; r < 8; ++r) {
      G [(size_t)(gr0 + r) * HD + n] = accG[nt][r] + bias;
      Mg[(size_t)(gr0 + r) * HD + n] = accM[nt][r];
    }
  }
}

// ---------------- H += Adj@Mg (per b,t slice) + BN stats atomics --------------
__global__ void __launch_bounds__(64)
k_gcn_combine(float* __restrict__ G, const float* __restrict__ Mg,
              const float* __restrict__ Adj,
              float* __restrict__ ssum, float* __restrict__ ssq) {
  __shared__ float Mgs[Vc * HD];
  __shared__ float Adjs[Vc * Vc];
  const int s = blockIdx.x;           // b*T + t
  const int t = s % Tc;
  const size_t base = (size_t)s * (Vc * HD);
  const int h = threadIdx.x;          // 0..63
  for (int i = h; i < Vc * HD; i += 64) Mgs[i] = Mg[base + i];
  for (int i = h; i < Vc * Vc; i += 64) Adjs[i] = Adj[i];
  __syncthreads();
  float sm = 0.f, sq = 0.f;
#pragma unroll 1
  for (int v = 0; v < Vc; ++v) {
    float acc = G[base + v * HD + h];
#pragma unroll
    for (int u = 0; u < Vc; ++u) acc += Adjs[v * Vc + u] * Mgs[u * HD + h];
    G[base + v * HD + h] = acc;
    sm += acc; sq += acc * acc;
  }
  atomicAdd(&ssum[t * HD + h], sm);   // BN over (B,V) per (t,h)
  atomicAdd(&ssq [t * HD + h], sq);
}

// ---------------- per-(t,h) BN normalize + ReLU, in place, float4 -------------
__global__ void k_gcn_norm(float* __restrict__ H, const float* __restrict__ ssum,
                           const float* __restrict__ ssq,
                           const float* __restrict__ gamma,
                           const float* __restrict__ beta) {
  size_t i4 = (size_t)blockIdx.x * blockDim.x + threadIdx.x;
  size_t n4 = (size_t)MROWS * HD / 4;
  if (i4 >= n4) return;
  size_t i = i4 * 4;
  int h = (int)(i % HD);
  int t = (int)((i / (Vc * HD)) % Tc);
  const float inv = 1.f / (Bc * Vc);
  float4 x = ((float4*)H)[i4];
  float xs[4] = {x.x, x.y, x.z, x.w};
  float o[4];
#pragma unroll
  for (int j = 0; j < 4; ++j) {
    float m = ssum[t * HD + h + j] * inv;
    float var = ssq[t * HD + h + j] * inv - m * m;
    float val = (xs[j] - m) * rsqrtf(var + EPS) * gamma[h + j] + beta[h + j];
    o[j] = val > 0.f ? val : 0.f;
  }
  ((float4*)H)[i4] = make_float4(o[0], o[1], o[2], o[3]);
}

// ---------------- temporal conv as tap-shifted WMMA GEMM ----------------------
// Y (B,T,Cin) -> Out (B,T,COUT); 3 taps, padding = dil. block: 8 waves x 16 rows.
template <int COUT>
__global__ void __launch_bounds__(256)
k_conv(const float* __restrict__ Y, int Cin,
       const float* __restrict__ Wc, const float* __restrict__ bc,
       int dil, float* __restrict__ Out) {
  constexpr int NT = COUT / 16;
  __shared__ v16bf As4[8 * 32];        // 128 rows x 32 k, fragment order (8 KB)
  __shared__ v16bf Bs4[NT * 32];       // 32 k x COUT, fragment order
  bhalf* As = (bhalf*)As4;
  bhalf* Bs = (bhalf*)Bs4;
  const int tid = threadIdx.x;
  const int b = blockIdx.y;
  const int t0 = blockIdx.x * 128;
  const int wave = tid >> 5, lane = tid & 31;
  f32x8 acc[NT] = {};

  for (int k = 0; k < 3; ++k) {
    const int toff = (k - 1) * dil;
    for (int kk = 0; kk < Cin; kk += 32) {
      __syncthreads();
      // stage A via float4 (global_load_b128); zero-pad t at the edges
      for (int i = tid; i < 128 * 8; i += 256) {
        int r = i >> 3, c4 = (i & 7) << 2;
        int t = t0 + r + toff;
        float4 x = make_float4(0.f, 0.f, 0.f, 0.f);
        if (t >= 0 && t < Tc) {
          const float* row = Y + ((size_t)b * Tc + t) * Cin + kk;
          x = ((const float4*)row)[c4 >> 2];
          if (kk + 32 < Cin)               // pull next k-chunk toward the caches
            __builtin_prefetch(row + 32 + c4, 0, 3);
        }
        store_pk4(As + a_perm_idx(r, c4, 1), x);
      }
      for (int i = tid; i < 32 * COUT; i += 256) {         // stage B: Wc[o,c,k]
        int c = i / COUT, o = i % COUT;
        Bs[b_perm_idx(c, o, NT)] = (bhalf)Wc[((size_t)o * Cin + (kk + c)) * 3 + k];
      }
      __syncthreads();
      v16bf a = As4[wave * 32 + lane];
#pragma unroll
      for (int nt = 0; nt < NT; ++nt) {
        v16bf bb = Bs4[nt * 32 + lane];
        acc[nt] = __builtin_amdgcn_wmma_f32_16x16x32_bf16(
            false, a, false, bb, (short)0, acc[nt], false, false);
      }
    }
  }

  const int nloc = lane & 15, half = lane >> 4;
  const int tr0 = t0 + wave * 16 + half * 8;
#pragma unroll
  for (int nt = 0; nt < NT; ++nt) {
    int o = nt * 16 + nloc;
    float bias = bc[o];
#pragma unroll
    for (int r = 0; r < 8; ++r)
      Out[((size_t)b * Tc + tr0 + r) * COUT + o] = acc[nt][r] + bias;
  }
}

// ---------------- per-channel BN stats / normalize for conv layers ------------
__global__ void __launch_bounds__(256)
k_chan_stats(const float* __restrict__ Y, int Cout, float* __restrict__ cs) {
  __shared__ float s1[256], s2[256];
  int o = blockIdx.x;
  float sm = 0.f, sq = 0.f;
  for (size_t i = threadIdx.x; i < (size_t)Bc * Tc; i += 256) {
    float v = Y[i * Cout + o];
    sm += v; sq += v * v;
  }
  s1[threadIdx.x] = sm; s2[threadIdx.x] = sq;
  __syncthreads();
  for (int st = 128; st > 0; st >>= 1) {
    if (threadIdx.x < st) {
      s1[threadIdx.x] += s1[threadIdx.x + st];
      s2[threadIdx.x] += s2[threadIdx.x + st];
    }
    __syncthreads();
  }
  if (threadIdx.x == 0) { cs[o] = s1[0]; cs[Cout + o] = s2[0]; }
}

__global__ void k_chan_norm(float* __restrict__ Y, int Cout,
                            const float* __restrict__ cs,
                            const float* __restrict__ gamma,
                            const float* __restrict__ beta) {
  size_t i4 = (size_t)blockIdx.x * blockDim.x + threadIdx.x;
  size_t n4 = (size_t)Bc * Tc * Cout / 4;
  if (i4 >= n4) return;
  size_t i = i4 * 4;
  int o = (int)(i % Cout);
  const float inv = 1.f / (Bc * Tc);
  float4 x = ((float4*)Y)[i4];
  float xs[4] = {x.x, x.y, x.z, x.w};
  float r[4];
#pragma unroll
  for (int j = 0; j < 4; ++j) {
    float m = cs[o + j] * inv;
    float var = cs[Cout + o + j] * inv - m * m;
    float val = (xs[j] - m) * rsqrtf(var + EPS) * gamma[o + j] + beta[o + j];
    r[j] = val > 0.f ? val : 0.f;
  }
  ((float4*)Y)[i4] = make_float4(r[0], r[1], r[2], r[3]);
}

// ---------------- mean over T + FC -------------------------------------------
__global__ void __launch_bounds__(128)
k_pool_fc(const float* __restrict__ Y2, const float* __restrict__ fcW,
          const float* __restrict__ fcb, float* __restrict__ out) {
  __shared__ float pooled[128];
  int b = blockIdx.x, o = threadIdx.x;
  float sm = 0.f;
  for (int t = 0; t < Tc; ++t) sm += Y2[((size_t)b * Tc + t) * 128 + o];
  pooled[o] = sm * (1.f / Tc);
  __syncthreads();
  if (o < NCc) {
    float acc = fcb[o];
#pragma unroll 4
    for (int c = 0; c < 128; ++c) acc += pooled[c] * fcW[c * NCc + o];
    out[b * NCc + o] = acc;
  }
}

// ---------------- host launcher ----------------------------------------------
extern "C" void kernel_launch(void* const* d_in, const int* in_sizes, int n_in,
                              void* d_out, int out_size, void* d_ws, size_t ws_size,
                              hipStream_t stream) {
  const float* X    = (const float*)d_in[0];   // (B,C,T,V)
  const int*   edge = (const int*)  d_in[1];   // (E,2)
  const float* W1  = (const float*)d_in[2];
  const float* s1W = (const float*)d_in[3];
  const float* s1b = (const float*)d_in[4];
  const float* g1  = (const float*)d_in[5];
  const float* b1  = (const float*)d_in[6];
  const float* W2  = (const float*)d_in[7];
  const float* s2W = (const float*)d_in[8];
  const float* s2b = (const float*)d_in[9];
  const float* g2  = (const float*)d_in[10];
  const float* b2  = (const float*)d_in[11];
  const float* c1W = (const float*)d_in[12];   // (64,1600,3)
  const float* c1b = (const float*)d_in[13];
  const float* tg1 = (const float*)d_in[14];
  const float* tb1 = (const float*)d_in[15];
  const float* c2W = (const float*)d_in[16];   // (128,64,3)
  const float* c2b = (const float*)d_in[17];
  const float* tg2 = (const float*)d_in[18];
  const float* tb2 = (const float*)d_in[19];
  const float* fcW = (const float*)d_in[20];   // (128,60)
  const float* fcb = (const float*)d_in[21];

  float* wsf    = (float*)d_ws;
  float* Adj    = wsf;                           // 1024 floats reserved
  float* stats  = wsf + 1024;                    // 2 * T*64 = 65536
  float* cstats = wsf + 1024 + 65536;            // 256
  float* bufA   = wsf + 1024 + 65536 + 256;      // 52,428,800 floats
  float* bufB   = bufA + (size_t)MROWS * HD;     // 52,428,800 floats
  float* out1   = bufB;                          // (B,T,64)  reuse bufB
  float* out2   = bufB + (size_t)16 * 1024 * 1024; // (B,T,128)

  const int gemmGrid = MROWS / 128;              // 6400
  const int normGrid = (MROWS * HD / 4 + 255) / 256;

  k_adj<<<1, 64, 0, stream>>>(edge, Adj);

  // ---- GCN layer 1 (K=3 padded to 32) ----
  k_zero<<<256, 256, 0, stream>>>(stats, 2 * Tc * HD);
  k_gcn_gemm<<<gemmGrid, 256, 0, stream>>>(X, 0, 3, 32, s1W, W1, s1b, bufA, bufB);
  k_gcn_combine<<<Bc * Tc, 64, 0, stream>>>(bufA, bufB, Adj, stats, stats + Tc * HD);
  k_gcn_norm<<<normGrid, 256, 0, stream>>>(bufA, stats, stats + Tc * HD, g1, b1);

  // ---- GCN layer 2 (K=64, in place on bufA) ----
  k_zero<<<256, 256, 0, stream>>>(stats, 2 * Tc * HD);
  k_gcn_gemm<<<gemmGrid, 256, 0, stream>>>(bufA, 1, 64, 64, s2W, W2, s2b, bufA, bufB);
  k_gcn_combine<<<Bc * Tc, 64, 0, stream>>>(bufA, bufB, Adj, stats, stats + Tc * HD);
  k_gcn_norm<<<normGrid, 256, 0, stream>>>(bufA, stats, stats + Tc * HD, g2, b2);

  // ---- temporal conv 1: (B,T,1600) -> (B,T,64), dil 1 ----
  k_conv<64><<<dim3(Tc / 128, Bc), 256, 0, stream>>>(bufA, Vc * HD, c1W, c1b, 1, out1);
  k_chan_stats<<<64, 256, 0, stream>>>(out1, 64, cstats);
  k_chan_norm<<<(Bc * Tc * 64 / 4 + 255) / 256, 256, 0, stream>>>(out1, 64, cstats, tg1, tb1);

  // ---- temporal conv 2: (B,T,64) -> (B,T,128), dil 2 ----
  k_conv<128><<<dim3(Tc / 128, Bc), 256, 0, stream>>>(out1, 64, c2W, c2b, 2, out2);
  k_chan_stats<<<128, 256, 0, stream>>>(out2, 128, cstats);
  k_chan_norm<<<(Bc * Tc * 128 / 4 + 255) / 256, 256, 0, stream>>>(out2, 128, cstats, tg2, tb2);

  // ---- mean over T + FC ----
  k_pool_fc<<<Bc, 128, 0, stream>>>(out2, fcW, fcb, (float*)d_out);
}
